// _GConv_2576980377710
// MI455X (gfx1250) — matrix operand order repaired and checked by reading, not compiled, not run on hardware
//
#include <hip/hip_runtime.h>
#include <hip/hip_bf16.h>

// Problem constants (match reference)
#define Bb 16
#define Hh 256
#define Ll 4096
#define KREV_LEN (Ll + 32)   // reversed filter + 32-zero tail (causal zeros)
#define NT 4                 // l-tiles register-blocked per wave in conv

typedef _Float16 v16h __attribute__((ext_vector_type(16)));
typedef _Float16 v8h  __attribute__((ext_vector_type(8)));
typedef float    v8f  __attribute__((ext_vector_type(8)));

// ---------------------------------------------------------------------------
// Stage 1: build normalized multiscale filter, store REVERSED + zero-padded:
//   krev[h][i] = k[h][L-1-i] (i < L), krev[h][L..L+31] = 0
// One block (256 threads) per h; LDS block reduction for the L2 norm.
// ---------------------------------------------------------------------------
__global__ __launch_bounds__(256) void build_k_kernel(
    const float* __restrict__ kern,   // [NS=8][1][H][KD=32]
    _Float16* __restrict__ krev)      // [H][KREV_LEN]
{
    __shared__ float kv[Ll];          // 16 KB
    __shared__ float red[256];
    const int h = blockIdx.x;

    float ss = 0.0f;
    for (int p = threadIdx.x; p < Ll; p += 256) {
        int i, off;
        if (p < 32) { i = 0; off = 0; }
        else        { i = (31 - __clz(p)) - 4; off = 1 << (i + 4); }
        const int   s  = (i == 0) ? 1 : (1 << (i - 1));
        const int   pl = p - off;
        float x = ((float)pl + 0.5f) / (float)s - 0.5f;
        x = fminf(fmaxf(x, 0.0f), 31.0f);
        int   x0 = (int)floorf(x);
        int   x1 = (x0 + 1 < 31) ? (x0 + 1) : 31;
        float w  = x - (float)x0;
        const float* kb = kern + ((size_t)i * Hh + h) * 32;
        float v = kb[x0] * (1.0f - w) + kb[x1] * w;
        v *= (float)(1 << (7 - i));   // multiplier == 2.0 for every h
        kv[p] = v;
        ss += v * v;
    }
    red[threadIdx.x] = ss;
    __syncthreads();
    for (int st = 128; st > 0; st >>= 1) {
        if ((int)threadIdx.x < st) red[threadIdx.x] += red[threadIdx.x + st];
        __syncthreads();
    }
    const float rn = rsqrtf(red[0]);

    _Float16* row = krev + (size_t)h * KREV_LEN;
    for (int p = threadIdx.x; p < Ll; p += 256)
        row[Ll - 1 - p] = (_Float16)(kv[p] * rn);
    if (threadIdx.x < 32)
        row[Ll + threadIdx.x] = (_Float16)0.0f;
}

// ---------------------------------------------------------------------------
// Stage 1b: u (B,H,L) f32 -> uh (H,B,L) f16 so the conv A-operand loads are
// aligned 32B vectors per lane.
// ---------------------------------------------------------------------------
__global__ __launch_bounds__(256) void u_to_half_kernel(
    const float* __restrict__ u, _Float16* __restrict__ uh)
{
    size_t idx = (size_t)blockIdx.x * blockDim.x + threadIdx.x;
    const size_t total = (size_t)Bb * Hh * Ll;
    if (idx >= total) return;
    int    l = (int)(idx % Ll);
    size_t t = idx / Ll;
    int    h = (int)(t % Hh);
    int    b = (int)(t / Hh);
    uh[((size_t)h * Bb + b) * Ll + l] = (_Float16)u[idx];
}

__global__ __launch_bounds__(256) void w_to_half_kernel(
    const float* __restrict__ W, _Float16* __restrict__ Wh)
{
    int idx = blockIdx.x * blockDim.x + threadIdx.x;
    if (idx < Hh * Hh) Wh[idx] = (_Float16)W[idx];
}

// ---------------------------------------------------------------------------
// Stage 2: causal Toeplitz conv via WMMA with swapped operand roles:
//   D[b][l] = sum_K  A[b][K] * B[K][l],  A = u (lag-contiguous, aligned),
//   B = Toeplitz slice loaded as ONE contiguous 32B read per lane from the
//   reversed zero-padded filter (padding supplies the causal zeros).
// A is independent of l, so each wave register-blocks NT=4 l-tiles and
// amortizes the A load over 4 WMMAs. No LDS, no gather.
// Block = 256 threads = 8 waves; grid (L/(16*NT*8)=8, H).
// Epilogue: + D[h]*u, exact GeLU, store f16 gh[B][L][H].
// ---------------------------------------------------------------------------
__global__ __launch_bounds__(256) void conv_wmma_kernel(
    const _Float16* __restrict__ krev,  // [H][KREV_LEN] reversed+padded
    const _Float16* __restrict__ uh,    // [H][B][L]
    const float*    __restrict__ u,     // [B][H][L]
    const float*    __restrict__ D,     // [H]
    _Float16*       __restrict__ gh)    // [B][L][H]
{
    const int lane = threadIdx.x & 31;
    const int wave = threadIdx.x >> 5;
    const int h    = blockIdx.y;
    const int l0   = (blockIdx.x * 8 + wave) * (16 * NT);  // base of NT l-tiles

    const int g = lane >> 4;   // half-select (shared by A and B layouts)
    const int n = lane & 15;   // A row = batch, B col = l offset

    v8f acc[NT] = {};

    const _Float16* arow = uh   + ((size_t)h * Bb + n) * Ll;
    const _Float16* krow = krev + (size_t)h * KREV_LEN;
    const int lmax = l0 + 16 * (NT - 1);

    for (int m0 = 0; m0 <= lmax; m0 += 32) {
        // A[b][K]: K_j = 16g + j  -> one aligned 32B load
        v16h a = *(const v16h*)(arow + m0 + 16 * g);

        // Prefetch next A tile (global_prefetch_b8)
        __builtin_prefetch(arow + m0 + 32, 0, 1);

        #pragma unroll
        for (int i = 0; i < NT; ++i) {
            const int li = l0 + 16 * i;
            if (m0 <= li) {
                // B[K][l]: value_j = k[(li+n) - (m0+16g+j)] = krev[base + j]
                const int base = Ll - 1 - li - n + m0 + 16 * g;
                v16h b;
                __builtin_memcpy(&b, krow + base, sizeof(b));  // unaligned 32B
                acc[i] = __builtin_amdgcn_wmma_f32_16x16x32_f16(
                    false, a, false, b, (short)0, acc[i], false, false);
            }
        }
    }

    // Epilogue: C/D layout -> acc[i][r] holds (b = r + 8g, l = l0 + 16i + n).
    const float Dh = D[h];
    #pragma unroll
    for (int i = 0; i < NT; ++i) {
        const int l = l0 + 16 * i + n;
        #pragma unroll
        for (int r = 0; r < 8; ++r) {
            const int bi = r + 8 * g;
            const float uv = u[((size_t)bi * Hh + h) * Ll + l];
            const float y  = acc[i][r] + Dh * uv;
            const float ge = 0.5f * y * (1.0f + erff(y * 0.70710678118654752f));
            gh[((size_t)bi * Ll + l) * Hh + h] = (_Float16)ge;
        }
    }
}

// ---------------------------------------------------------------------------
// Stage 3: output projection out[b,ho,l] = sum_hin W[ho,hin]*g[b,l,hin]+b_out.
// Block = 128 threads = 4 waves; wave owns one (ho-tile, l-tile).
// Grid: (H/16, L/64, B). K-loop: 8 x 32 over hin, fully unrolled -> 8 WMMAs.
// ---------------------------------------------------------------------------
__global__ __launch_bounds__(128) void proj_wmma_kernel(
    const _Float16* __restrict__ Wh,    // [H][H]
    const _Float16* __restrict__ gh,    // [B][L][H]
    const float*    __restrict__ bout,  // [H]
    float*          __restrict__ out)   // [B][H][L]
{
    const int lane = threadIdx.x & 31;
    const int wave = threadIdx.x >> 5;
    const int ho0  = blockIdx.x * 16;
    const int l0   = (blockIdx.y * 4 + wave) * 16;
    const int bi   = blockIdx.z;

    const int g = lane >> 4;
    const int M = lane & 15;   // A row = ho offset; B col = l offset

    v8f acc = {};
    const _Float16* wrow = Wh + (size_t)(ho0 + M) * Hh;
    const _Float16* grow = gh + ((size_t)bi * Ll + (l0 + M)) * Hh;

    #pragma unroll
    for (int k0 = 0; k0 < Hh; k0 += 32) {
        v16h a = *(const v16h*)(wrow + k0 + 16 * g);

        v16h b = *(const v16h*)(grow + k0 + 16 * g);  // 32B aligned

        acc = __builtin_amdgcn_wmma_f32_16x16x32_f16(
            false, a, false, b, (short)0, acc, false, false);
    }

    #pragma unroll
    for (int r = 0; r < 8; ++r) {
        const int ho = ho0 + r + 8 * g;
        out[((size_t)bi * Hh + ho) * Ll + l0 + M] = acc[r] + bout[ho];
    }
}

// ---------------------------------------------------------------------------
// Host launcher
// ---------------------------------------------------------------------------
extern "C" void kernel_launch(void* const* d_in, const int* in_sizes, int n_in,
                              void* d_out, int out_size, void* d_ws, size_t ws_size,
                              hipStream_t stream)
{
    const float* u     = (const float*)d_in[0];   // (B,H,L)
    const float* kern  = (const float*)d_in[1];   // (8,1,H,32)
    const float* Dm    = (const float*)d_in[2];   // (1,H)
    const float* W_out = (const float*)d_in[3];   // (H,H)
    const float* b_out = (const float*)d_in[4];   // (H,)
    float* out = (float*)d_out;                   // (B,H,L)

    // Workspace layout (halves); offsets are 256B-aligned by construction.
    char* wsb = (char*)d_ws;
    const size_t KREV_B = (size_t)Hh * KREV_LEN * 2;   // ~2 MB (8256B rows)
    const size_t UH_B   = (size_t)Hh * Bb * Ll * 2;    // 32 MB
    const size_t WH_B   = (size_t)Hh * Hh * 2;         // 128 KB
    _Float16* krev = (_Float16*)(wsb);
    _Float16* uh   = (_Float16*)(wsb + KREV_B);
    _Float16* Wh   = (_Float16*)(wsb + KREV_B + UH_B);
    _Float16* gh   = (_Float16*)(wsb + KREV_B + UH_B + WH_B);  // 32 MB

    // Stage 1: filter build + normalize + reverse/pad
    build_k_kernel<<<dim3(Hh), dim3(256), 0, stream>>>(kern, krev);

    // Stage 1b: precision/layout conversions
    {
        const size_t total = (size_t)Bb * Hh * Ll;
        u_to_half_kernel<<<dim3((unsigned)((total + 255) / 256)), dim3(256), 0, stream>>>(u, uh);
        w_to_half_kernel<<<dim3((Hh * Hh + 255) / 256), dim3(256), 0, stream>>>(W_out, Wh);
    }

    // Stage 2: causal Toeplitz conv (WMMA, NT-blocked) + skip + GeLU
    conv_wmma_kernel<<<dim3(Ll / (16 * NT * 8), Hh), dim3(256), 0, stream>>>(krev, uh, u, Dm, gh);

    // Stage 3: output projection (WMMA) + bias
    proj_wmma_kernel<<<dim3(Hh / 16, Ll / 64, Bb), dim3(128), 0, stream>>>(Wh, gh, b_out, out);

    (void)in_sizes; (void)n_in; (void)out_size; (void)ws_size;
}
